// MechanismDisentanglementModule_30734785970928
// MI455X (gfx1250) — matrix operand order repaired and checked by reading, not compiled, not run on hardware
//
#include <hip/hip_runtime.h>
#include <hip/hip_bf16.h>

// ---------------------------------------------------------------------------
// MechanismDisentanglementModule — fused batched-expert MLP for MI455X gfx1250
//   B=131072 D=256 H=64 E=20 O=3 C=5
//
// GEMM1 (B x 256 x 64, 20 experts): v_wmma_f32_16x16x32_bf16, z tile held in
//   registers for all 20 experts; W1[e] bf16 staged to LDS.
// GEMM2 (B x 64 x 3): WMMA against a zero-padded 16-col W2 B-image.
// NEW: weights pre-converted to bf16 images in d_ws, staged per expert with
//   double-buffered GLOBAL_LOAD_ASYNC_TO_LDS_B128 (ASYNCcnt) — staging fully
//   overlapped with WMMA compute, one barrier per expert.
// ---------------------------------------------------------------------------

#define BATCH   131072
#define DIMD    256
#define DIMH    64
#define NEXP    20
#define NOUT    3
#define NCAT    5
#define LN_EPS  1e-5f

#define ROWS_PER_BLOCK 256      // 16 waves * 16 rows
#define THREADS        512
#define NWAVES         16
#define W1_LDS_STRIDE  264      // 256 + 8 pad (bf16 elems) -> 528B row, 16B aligned
#define A_LDS_STRIDE   72       // 64 + 8 pad (bf16 elems)  -> 144B row, 16B aligned

#define W1_WS_ELEMS    (NEXP * DIMH * DIMD)   // 327680 bf16
#define W2_WS_ELEMS    (NEXP * 16 * DIMH)     // 20480 bf16 (zero-padded image)

typedef __attribute__((ext_vector_type(16))) __bf16 v16bf;
typedef __attribute__((ext_vector_type(8)))  float  v8f;

union FragBF {
    v16bf v;
    unsigned u[8];
    uint4    q[2];
};

// round-to-nearest-even f32 -> bf16, packed pair
__device__ __forceinline__ unsigned pk_bf16(float a, float b) {
    unsigned ua = __float_as_uint(a);
    unsigned ub = __float_as_uint(b);
    ua = (ua + 0x7FFFu + ((ua >> 16) & 1u)) >> 16;
    ub = (ub + 0x7FFFu + ((ub >> 16) & 1u)) >> 16;
    return ua | (ub << 16);
}

__device__ __forceinline__ unsigned short bf16_rne(float v) {
    unsigned u = __float_as_uint(v);
    return (unsigned short)((u + 0x7FFFu + ((u >> 16) & 1u)) >> 16);
}

// ---------------------------------------------------------------------------
// Kernel 0: build bf16 weight images in workspace.
//   wsW1 [E][H][D]   : W1 transposed (B-matrix layout feed, k-innermost)
//   wsW2 [E][16][D=64]: W2 transposed o-major, rows o=3..15 zeroed
// ---------------------------------------------------------------------------
__global__ __launch_bounds__(256)
void mdm_pack_weights(const float* __restrict__ W1, const float* __restrict__ W2,
                      unsigned short* __restrict__ wsW1,
                      unsigned short* __restrict__ wsW2) {
    int tid = blockIdx.x * 256 + threadIdx.x;
    if (tid < W1_WS_ELEMS) {
        int k   = tid & (DIMD - 1);
        int rem = tid >> 8;
        int h   = rem & (DIMH - 1);
        int e   = rem >> 6;
        wsW1[(e * DIMH + h) * DIMD + k] = bf16_rne(W1[(e * DIMD + k) * DIMH + h]);
    } else if (tid < W1_WS_ELEMS + W2_WS_ELEMS) {
        int idx = tid - W1_WS_ELEMS;
        int e = idx >> 10;          // / 1024
        int r = idx & 1023;
        int o = r >> 6;
        int h = r & 63;
        unsigned short v = 0;
        if (o < NOUT) v = bf16_rne(W2[(e * DIMH + h) * NOUT + o]);
        wsW2[idx] = v;
    }
}

// ---------------------------------------------------------------------------
// Kernel 1: fused expert MLP + LN + SiLU + WMMA head + sigmoid + cat argmax
// ---------------------------------------------------------------------------
__global__ __launch_bounds__(THREADS)
void mdm_main(const float* __restrict__ z,
              const int*   __restrict__ cat_ids,
              const float* __restrict__ b1,
              const float* __restrict__ gamma,
              const float* __restrict__ beta,
              const float* __restrict__ b2,
              const unsigned short* __restrict__ wsW1,
              const unsigned short* __restrict__ wsW2,
              float* __restrict__ out,      // [B][E][3]
              float* __restrict__ probs,    // [B][E]
              float* __restrict__ catlog,   // [B][C]
              float* __restrict__ dom)      // [B]
{
    __shared__ __align__(16) unsigned short sW1d[2][DIMH * W1_LDS_STRIDE];     // 66 KB
    __shared__ __align__(16) unsigned short sW2d[2][16 * DIMH];                //  4 KB
    __shared__ __align__(16) unsigned short sA[NWAVES * 16 * A_LDS_STRIDE];    // 36 KB
    __shared__ float sProbs[ROWS_PER_BLOCK * NEXP];                            // 20 KB

    const int tid  = threadIdx.x;
    const int wave = tid >> 5;
    const int lane = tid & 31;
    const int n0   = lane & 15;
    const int hi   = lane >> 4;           // which 16-lane half

    const int blockRow = blockIdx.x * ROWS_PER_BLOCK;
    const int waveRow  = blockRow + wave * 16;
    const int outRow   = waveRow + hi * 8;    // C-matrix: VGPR r -> row r (+8 upper half)

    unsigned short* myA = &sA[wave * 16 * A_LDS_STRIDE];

    // async prefetch of expert en's weights into LDS buffer en&1
    auto prefetch = [&](int en) {
        int buf = en & 1;
        {   // W1: 512 threads x 64B = 32 KB
            int h  = tid >> 3;
            int ks = (tid & 7) * 32;
            unsigned long long g =
                (unsigned long long)(wsW1 + ((size_t)en * DIMH + h) * DIMD + ks);
            unsigned l = (unsigned)(size_t)(&sW1d[buf][h * W1_LDS_STRIDE + ks]);
#pragma unroll
            for (int i = 0; i < 4; ++i) {
                asm volatile("global_load_async_to_lds_b128 %0, %1, off"
                             :: "v"(l + i * 16), "v"(g + i * 16) : "memory");
            }
        }
        if (tid < 128) {   // W2 image: 128 threads x 16B = 2 KB
            unsigned long long g =
                (unsigned long long)(wsW2 + (size_t)en * (16 * DIMH) + tid * 8);
            unsigned l = (unsigned)(size_t)(&sW2d[buf][tid * 8]);
            asm volatile("global_load_async_to_lds_b128 %0, %1, off"
                         :: "v"(l), "v"(g) : "memory");
        }
    };

    // ---- Load this wave's 16x256 A tile once, convert to bf16 fragments ----
    // A 16x32 bf16 layout: lanes 0-15 M=lane,  K in {kb..kb+7, kb+16..kb+23}, kb=0
    //                      lanes16-31 M=lane-16, kb=8
    prefetch(0);
    FragBF afrag[8];
    {
        const float* zr = z + (size_t)(waveRow + n0) * DIMD;
#pragma unroll
        for (int kc = 0; kc < 8; ++kc) {
            int kb = kc * 32 + hi * 8;
            float4 f0 = *(const float4*)(zr + kb);
            float4 f1 = *(const float4*)(zr + kb + 4);
            float4 f2 = *(const float4*)(zr + kb + 16);
            float4 f3 = *(const float4*)(zr + kb + 20);
            afrag[kc].u[0] = pk_bf16(f0.x, f0.y);
            afrag[kc].u[1] = pk_bf16(f0.z, f0.w);
            afrag[kc].u[2] = pk_bf16(f1.x, f1.y);
            afrag[kc].u[3] = pk_bf16(f1.z, f1.w);
            afrag[kc].u[4] = pk_bf16(f2.x, f2.y);
            afrag[kc].u[5] = pk_bf16(f2.z, f2.w);
            afrag[kc].u[6] = pk_bf16(f3.x, f3.y);
            afrag[kc].u[7] = pk_bf16(f3.z, f3.w);
        }
    }

    for (int e = 0; e < NEXP; ++e) {
        // my async loads done -> barrier: buffer e&1 complete for all waves,
        // and everyone is past expert e-1, so buffer (e+1)&1 is reusable.
        asm volatile("s_wait_asynccnt 0x0" ::: "memory");
        __syncthreads();
        if (e + 1 < NEXP) prefetch(e + 1);

        const unsigned short* cW1 = sW1d[e & 1];
        const unsigned short* cW2 = sW2d[e & 1];

        // per-lane LN params (L2-resident broadcast loads)
        float bb[4], gg[4], be[4];
#pragma unroll
        for (int nt = 0; nt < 4; ++nt) {
            int idx = e * DIMH + nt * 16 + n0;
            bb[nt] = b1[idx];
            gg[nt] = gamma[idx];
            be[nt] = beta[idx];
        }

        // ---- GEMM1: 16x64 = (16x256)x(256x64) via 8x4 WMMA bf16 ----
        v8f acc[4];
#pragma unroll
        for (int nt = 0; nt < 4; ++nt) acc[nt] = (v8f)0.0f;

#pragma unroll
        for (int kc = 0; kc < 8; ++kc) {
#pragma unroll
            for (int nt = 0; nt < 4; ++nt) {
                // B 32x16: lanes 0-15 K={0-7,16-23}, lanes 16-31 K={8-15,24-31}
                FragBF bf;
                const uint4* p = (const uint4*)
                    (&cW1[(nt * 16 + n0) * W1_LDS_STRIDE + kc * 32 + hi * 8]);
                bf.q[0] = p[0];
                bf.q[1] = p[2];
                acc[nt] = __builtin_amdgcn_wmma_f32_16x16x32_bf16(
                    false, afrag[kc].v, false, bf.v, (short)0, acc[nt],
                    false, false);
            }
        }

        // ---- + b1, LayerNorm stats (single pass E[x], E[x^2]) ----
        float s1[8], s2[8];
#pragma unroll
        for (int r = 0; r < 8; ++r) {
            float sum = 0.f, sq = 0.f;
#pragma unroll
            for (int nt = 0; nt < 4; ++nt) {
                float a0 = acc[nt][r] + bb[nt];
                acc[nt][r] = a0;
                sum += a0;
                sq  += a0 * a0;
            }
            s1[r] = sum; s2[r] = sq;
        }
#pragma unroll
        for (int m = 1; m <= 8; m <<= 1) {   // reduce over the 16-lane half
#pragma unroll
            for (int r = 0; r < 8; ++r) {
                s1[r] += __shfl_xor(s1[r], m, 32);
                s2[r] += __shfl_xor(s2[r], m, 32);
            }
        }

        // ---- normalize + SiLU, write activated tile a[m][h] to LDS ----
#pragma unroll
        for (int r = 0; r < 8; ++r) {
            float mu  = s1[r] * (1.0f / DIMH);
            float var = s2[r] * (1.0f / DIMH) - mu * mu;
            float rs  = rsqrtf(var + LN_EPS);
            int   mrow = hi * 8 + r;
#pragma unroll
            for (int nt = 0; nt < 4; ++nt) {
                float x = (acc[nt][r] - mu) * rs * gg[nt] + be[nt];
                float a = x * __frcp_rn(1.0f + __expf(-x));   // SiLU
                myA[mrow * A_LDS_STRIDE + nt * 16 + n0] =
                    (unsigned short)(__float_as_uint(a) >> 16);  // -> ds_store_b16_d16_hi
            }
        }

        // ---- GEMM2: (16x64)x(64x16-padded) via 2 WMMA, cols 0..2 valid ----
        v8f acc2 = (v8f)0.0f;
#pragma unroll
        for (int kc = 0; kc < 2; ++kc) {
            int kb = kc * 32 + hi * 8;
            FragBF a2, b2f;
            a2.q[0] = *(const uint4*)(&myA[n0 * A_LDS_STRIDE + kb]);
            a2.q[1] = *(const uint4*)(&myA[n0 * A_LDS_STRIDE + kb + 16]);
            b2f.q[0] = *(const uint4*)(&cW2[n0 * DIMH + kc * 32 + hi * 8]);
            b2f.q[1] = *(const uint4*)(&cW2[n0 * DIMH + kc * 32 + hi * 8 + 16]);
            acc2 = __builtin_amdgcn_wmma_f32_16x16x32_bf16(
                false, a2.v, false, b2f.v, (short)0, acc2, false, false);
        }

        // ---- writers: lanes n0<3 own output column o=n0 (C layout) ----
        if (n0 < NOUT) {
            float b2s = b2[e * NOUT + n0];
#pragma unroll
            for (int r = 0; r < 8; ++r) {
                size_t row = (size_t)(outRow + r);
                float o   = acc2[r] + b2s;
                out[(row * NEXP + e) * NOUT + n0] = o;
                if (n0 == 0) {
                    float p = __frcp_rn(1.0f + __expf(-o));
                    probs[row * NEXP + e] = p;
                    sProbs[(wave * 16 + hi * 8 + r) * NEXP + e] = p;
                }
            }
        }
    }

    // ---- per-row category sums + argmax ----
    __syncthreads();
    if (tid < ROWS_PER_BLOCK) {
        size_t row = (size_t)(blockRow + tid);
        float pr[NEXP];
#pragma unroll
        for (int e = 0; e < NEXP; ++e) pr[e] = sProbs[tid * NEXP + e];
        float cats[NCAT];
#pragma unroll
        for (int c = 0; c < NCAT; ++c) {
            float s = 0.f;
#pragma unroll
            for (int e = 0; e < NEXP; ++e)
                s += (cat_ids[e] == c) ? pr[e] : 0.f;
            cats[c] = s;
            catlog[row * NCAT + c] = s;
        }
        int best = 0; float bv = cats[0];
#pragma unroll
        for (int c = 1; c < NCAT; ++c)
            if (cats[c] > bv) { bv = cats[c]; best = c; }   // first-max = argmax
        dom[row] = (float)best;
    }
}

// ---------------------------------------------------------------------------
extern "C" void kernel_launch(void* const* d_in, const int* in_sizes, int n_in,
                              void* d_out, int out_size, void* d_ws, size_t ws_size,
                              hipStream_t stream) {
    const float* z       = (const float*)d_in[0];
    const int*   cat_ids = (const int*)  d_in[1];
    const float* W1      = (const float*)d_in[2];
    const float* b1      = (const float*)d_in[3];
    const float* gamma   = (const float*)d_in[4];
    const float* beta    = (const float*)d_in[5];
    const float* W2      = (const float*)d_in[6];
    const float* b2      = (const float*)d_in[7];

    unsigned short* wsW1 = (unsigned short*)d_ws;        // 640 KiB
    unsigned short* wsW2 = wsW1 + W1_WS_ELEMS;           //  40 KiB (padded image)

    float* out    = (float*)d_out;                       // [B][E][3]
    float* probs  = out    + (size_t)BATCH * NEXP * NOUT;
    float* catlog = probs  + (size_t)BATCH * NEXP;
    float* dom    = catlog + (size_t)BATCH * NCAT;

    // Kernel 0: pack both weight images as bf16
    {
        int total  = W1_WS_ELEMS + W2_WS_ELEMS;
        int blocks = (total + 255) / 256;
        mdm_pack_weights<<<blocks, 256, 0, stream>>>(W1, W2, wsW1, wsW2);
    }
    // Kernel 1: fused pipeline
    {
        int blocks = BATCH / ROWS_PER_BLOCK;   // 512
        mdm_main<<<blocks, THREADS, 0, stream>>>(z, cat_ids, b1, gamma, beta,
                                                 b2, wsW1, wsW2,
                                                 out, probs, catlog, dom);
    }
}
